// ResidualAttentionBlock_78769700208641
// MI455X (gfx1250) — matrix operand order repaired and checked by reading, not compiled, not run on hardware
//
#include <hip/hip_runtime.h>
#include <math.h>

typedef unsigned short u16;
typedef __attribute__((ext_vector_type(16))) __bf16 v16bf;
typedef __attribute__((ext_vector_type(8)))  __bf16 v8bf;
typedef __attribute__((ext_vector_type(8)))  float  v8f;

union Frag { v16bf v; v8bf h[2]; };

// ---- gfx1250 async global->LDS path (compile-safe feature detection) ----
#if defined(__HIP_DEVICE_COMPILE__) && defined(__gfx1250__) && \
    __has_builtin(__builtin_amdgcn_global_load_async_to_lds_b128) && \
    __has_builtin(__builtin_amdgcn_s_wait_asynccnt)
#define USE_ASYNC_LDS 1
#else
#define USE_ASYNC_LDS 0
#endif

#if USE_ASYNC_LDS
typedef int v4i_t __attribute__((vector_size(16)));
typedef __attribute__((address_space(1))) v4i_t* g4ptr;  // global (AS1) int4*
typedef __attribute__((address_space(3))) v4i_t* l4ptr;  // LDS (AS3) int4*
__device__ inline void async_ld16(const u16* g, u16* l) {
  __builtin_amdgcn_global_load_async_to_lds_b128((g4ptr)g, (l4ptr)l, 0, 0);
}
#endif

__device__ inline u16 f2bf(float f) {
  unsigned u = __float_as_uint(f);
  unsigned r = u + 0x7FFFu + ((u >> 16) & 1u);  // round-to-nearest-even
  return (u16)(r >> 16);
}

__device__ inline v8f zero8() {
  v8f z;
#pragma unroll
  for (int i = 0; i < 8; ++i) z[i] = 0.0f;
  return z;
}

__device__ inline v8f wmma_bf16(v16bf a, v16bf b, v8f c) {
  return __builtin_amdgcn_wmma_f32_16x16x32_bf16(
      false, a, false, b, (short)0, c, false, false);
}

__device__ inline float gelu_exact(float x) {
  return 0.5f * x * (1.0f + erff(x * 0.70710678118654752440f));
}

__device__ inline float rmax16(float v) {
#pragma unroll
  for (int m = 8; m >= 1; m >>= 1) v = fmaxf(v, __shfl_xor(v, m, 16));
  return v;
}
__device__ inline float rsum16(float v) {
#pragma unroll
  for (int m = 8; m >= 1; m >>= 1) v += __shfl_xor(v, m, 16);
  return v;
}

// ---------------- elementwise fp32 -> bf16 ----------------
__global__ __launch_bounds__(256) void cvt_bf16_kernel(const float* __restrict__ src,
                                                       u16* __restrict__ dst, size_t n) {
  size_t i = (size_t)blockIdx.x * 256 + threadIdx.x;
  if (i < n) dst[i] = f2bf(src[i]);
}

// ---------------- LayerNorm (fp32 stats) -> bf16, C == 1024 ----------------
__global__ __launch_bounds__(256) void ln_bf16_kernel(const float* __restrict__ x,
                                                      const float* __restrict__ w,
                                                      const float* __restrict__ b,
                                                      u16* __restrict__ out, int C) {
  int row = blockIdx.x;
  int tid = threadIdx.x;
  const float4 v = ((const float4*)(x + (size_t)row * C))[tid];
  float s  = v.x + v.y + v.z + v.w;
  float s2 = v.x * v.x + v.y * v.y + v.z * v.z + v.w * v.w;
#pragma unroll
  for (int m = 16; m >= 1; m >>= 1) {
    s  += __shfl_xor(s, m, 32);
    s2 += __shfl_xor(s2, m, 32);
  }
  __shared__ float red[16];
  int wid = tid >> 5, lane = tid & 31;
  if (lane == 0) { red[wid] = s; red[8 + wid] = s2; }
  __syncthreads();
  float ts = 0.f, ts2 = 0.f;
#pragma unroll
  for (int i = 0; i < 8; ++i) { ts += red[i]; ts2 += red[8 + i]; }
  float mu  = ts / (float)C;
  float inv = rsqrtf(ts2 / (float)C - mu * mu + 1e-5f);
  int base = tid * 4;
  u16* o = out + (size_t)row * C + base;
  o[0] = f2bf((v.x - mu) * inv * w[base + 0] + b[base + 0]);
  o[1] = f2bf((v.y - mu) * inv * w[base + 1] + b[base + 1]);
  o[2] = f2bf((v.z - mu) * inv * w[base + 2] + b[base + 2]);
  o[3] = f2bf((v.w - mu) * inv * w[base + 3] + b[base + 3]);
}

// ---------------- V transpose: [b*S+s, h*D+d] -> [b,h,d, s(padded Sp)] ----------------
__global__ __launch_bounds__(256) void vtrans_kernel(const u16* __restrict__ v,
                                                     u16* __restrict__ vT,
                                                     int B, int S, int H, int D, int Sp) {
  size_t idx = (size_t)blockIdx.x * 256 + threadIdx.x;
  size_t total = (size_t)B * H * D * Sp;
  if (idx >= total) return;
  int s = (int)(idx % Sp);
  size_t rest = idx / Sp;
  int d = (int)(rest % D); rest /= D;
  int h = (int)(rest % H);
  int b = (int)(rest / H);
  u16 val = 0;
  if (s < S) val = v[(size_t)(b * S + s) * (H * D) + h * D + d];
  vT[idx] = val;
}

// ---------------- bf16 WMMA GEMM:  out = epi( A[M,K] * W[N,K]^T ) ----------------
// Block 128(M) x 128(N), 256 threads = 8 waves, wave tile 32x64 (2x4 WMMA), K-step 32.
// Double-buffered LDS, one barrier/iter, branch-free inner loop (row-clamped A).
// Requires: N % 128 == 0, K % 32 == 0.
__global__ __launch_bounds__(256) void gemm_bf16_kernel(
    const u16* __restrict__ A, const u16* __restrict__ W,
    const float* __restrict__ bias, const float* __restrict__ resid,
    float* __restrict__ outF, u16* __restrict__ outB,
    int M, int N, int K, float alpha, int act) {
  const int LDT = 40;  // padded row stride in u16 (80B: conflict-free b128 reads)
  __shared__ __align__(16) u16 As[2][128 * 40];
  __shared__ __align__(16) u16 Bs[2][128 * 40];
  int tid = threadIdx.x;
  int wid = tid >> 5, lane = tid & 31, half = lane >> 4, l16 = lane & 15;
  int wm = (wid & 3) * 32;   // wave M offset in block
  int wn = (wid >> 2) * 64;  // wave N offset in block
  int m0 = blockIdx.y * 128, n0 = blockIdx.x * 128;

  v8f c[2][4];
#pragma unroll
  for (int mi = 0; mi < 2; ++mi)
#pragma unroll
    for (int ni = 0; ni < 4; ++ni) c[mi][ni] = zero8();

  // per-thread tile-copy coordinates: 128 rows x 32 cols, 16 u16 per thread
  int trow = tid >> 1, tcol = (tid & 1) * 16;
  int ra = m0 + trow;
  ra = ra < M ? ra : (M - 1);  // clamp: garbage only reaches unstored output rows
  const u16* gA = A + (size_t)ra * K + tcol;
  const u16* gB = W + (size_t)(n0 + trow) * K + tcol;
  int sOff = trow * LDT + tcol;
  int nIt = K >> 5;

#if USE_ASYNC_LDS
  // ---- ASYNCcnt pipeline: wait tile i; barrier; issue tile i+1; compute tile i ----
  async_ld16(gA,     &As[0][sOff]);
  async_ld16(gA + 8, &As[0][sOff + 8]);
  async_ld16(gB,     &Bs[0][sOff]);
  async_ld16(gB + 8, &Bs[0][sOff + 8]);
  for (int i = 0; i < nIt; ++i) {
    int cb = i & 1;
    __builtin_amdgcn_s_wait_asynccnt(0);
    __syncthreads();
    if (i + 1 < nIt) {
      int nb = cb ^ 1;
      const u16* pa = gA + (size_t)(i + 1) * 32;
      const u16* pb = gB + (size_t)(i + 1) * 32;
      async_ld16(pa,     &As[nb][sOff]);
      async_ld16(pa + 8, &As[nb][sOff + 8]);
      async_ld16(pb,     &Bs[nb][sOff]);
      async_ld16(pb + 8, &Bs[nb][sOff + 8]);
    }
    Frag fa[2], fb[4];
#pragma unroll
    for (int mi = 0; mi < 2; ++mi) {
      const u16* ap = &As[cb][(wm + mi * 16 + l16) * LDT + half * 8];
      fa[mi].h[0] = *(const v8bf*)ap;
      fa[mi].h[1] = *(const v8bf*)(ap + 16);
    }
#pragma unroll
    for (int ni = 0; ni < 4; ++ni) {
      const u16* bp = &Bs[cb][(wn + ni * 16 + l16) * LDT + half * 8];
      fb[ni].h[0] = *(const v8bf*)bp;
      fb[ni].h[1] = *(const v8bf*)(bp + 16);
    }
#pragma unroll
    for (int mi = 0; mi < 2; ++mi)
#pragma unroll
      for (int ni = 0; ni < 4; ++ni)
        c[mi][ni] = wmma_bf16(fa[mi].v, fb[ni].v, c[mi][ni]);
  }
#else
  // ---- register-staged double buffer: store tile i; barrier; load tile i+1; compute tile i ----
  uint4 a0 = *(const uint4*)gA;
  uint4 a1 = *(const uint4*)(gA + 8);
  uint4 b0 = *(const uint4*)gB;
  uint4 b1 = *(const uint4*)(gB + 8);
  for (int i = 0; i < nIt; ++i) {
    int cb = i & 1;
    *(uint4*)&As[cb][sOff]     = a0;
    *(uint4*)&As[cb][sOff + 8] = a1;
    *(uint4*)&Bs[cb][sOff]     = b0;
    *(uint4*)&Bs[cb][sOff + 8] = b1;
    __syncthreads();
    if (i + 1 < nIt) {
      const u16* pa = gA + (size_t)(i + 1) * 32;
      const u16* pb = gB + (size_t)(i + 1) * 32;
      a0 = *(const uint4*)pa;
      a1 = *(const uint4*)(pa + 8);
      b0 = *(const uint4*)pb;
      b1 = *(const uint4*)(pb + 8);
    }
    Frag fa[2], fb[4];
#pragma unroll
    for (int mi = 0; mi < 2; ++mi) {
      const u16* ap = &As[cb][(wm + mi * 16 + l16) * LDT + half * 8];
      fa[mi].h[0] = *(const v8bf*)ap;
      fa[mi].h[1] = *(const v8bf*)(ap + 16);
    }
#pragma unroll
    for (int ni = 0; ni < 4; ++ni) {
      const u16* bp = &Bs[cb][(wn + ni * 16 + l16) * LDT + half * 8];
      fb[ni].h[0] = *(const v8bf*)bp;
      fb[ni].h[1] = *(const v8bf*)(bp + 16);
    }
#pragma unroll
    for (int mi = 0; mi < 2; ++mi)
#pragma unroll
      for (int ni = 0; ni < 4; ++ni)
        c[mi][ni] = wmma_bf16(fa[mi].v, fb[ni].v, c[mi][ni]);
  }
#endif

  // ---- epilogue: v = (acc + bias) * alpha ; gelu? ; + resid ; store f32 / bf16 ----
#pragma unroll
  for (int mi = 0; mi < 2; ++mi)
#pragma unroll
    for (int ni = 0; ni < 4; ++ni)
#pragma unroll
      for (int r = 0; r < 8; ++r) {
        int gm = m0 + wm + mi * 16 + r + half * 8;
        int gn = n0 + wn + ni * 16 + l16;
        if (gm < M) {
          float v = c[mi][ni][r];
          if (bias) v += bias[gn];
          v *= alpha;
          if (act) v = gelu_exact(v);
          size_t off = (size_t)gm * N + gn;
          if (resid) v += resid[off];
          if (outF) outF[off] = v;
          if (outB) outB[off] = f2bf(v);
        }
      }
}

// ---------------- Flash attention (per-wave 16-row q tile, k chunks of 32) ----------------
__global__ __launch_bounds__(256) void flash_attn_kernel(
    const u16* __restrict__ q, const u16* __restrict__ k,
    const u16* __restrict__ vT, u16* __restrict__ o,
    int B, int S, int H, int D, int Sp) {
  const int LDP = 40;
  __shared__ __align__(16) u16 Pl[8 * 16 * 40];
  int tid = threadIdx.x, wid = tid >> 5, lane = tid & 31, half = lane >> 4, l16 = lane & 15;
  int nQT = (S + 15) / 16;
  int qt = blockIdx.x * 8 + wid;
  if (qt >= nQT) return;
  int h = blockIdx.y, b = blockIdx.z;
  int C = H * D;
  int q0 = qt * 16;

  Frag fq[2];
  {
    size_t qrow = (size_t)(b * S + q0 + l16) * C + h * D;
#pragma unroll
    for (int dc = 0; dc < 2; ++dc) {
      const u16* p = q + qrow + dc * 32 + half * 8;
      fq[dc].h[0] = *(const v8bf*)p;
      fq[dc].h[1] = *(const v8bf*)(p + 16);
    }
  }

  v8f oacc[4];
#pragma unroll
  for (int nt = 0; nt < 4; ++nt) oacc[nt] = zero8();
  float mrun[8], lrun[8];
#pragma unroll
  for (int r = 0; r < 8; ++r) { mrun[r] = -1e30f; lrun[r] = 0.0f; }

  u16* pw = &Pl[wid * 16 * LDP];

  for (int kc = 0; kc < S; kc += 32) {
    v8f sa0 = zero8(), sa1 = zero8();
#pragma unroll
    for (int t = 0; t < 2; ++t) {
      size_t krow = (size_t)(b * S + kc + t * 16 + l16) * C + h * D;
#pragma unroll
      for (int dc = 0; dc < 2; ++dc) {
        Frag fk;
        const u16* p = k + krow + dc * 32 + half * 8;
        fk.h[0] = *(const v8bf*)p;
        fk.h[1] = *(const v8bf*)(p + 16);
        if (t == 0) sa0 = wmma_bf16(fq[dc].v, fk.v, sa0);
        else        sa1 = wmma_bf16(fq[dc].v, fk.v, sa1);
      }
    }
    bool msk0 = (kc + l16) >= S;
    bool msk1 = (kc + 16 + l16) >= S;
#pragma unroll
    for (int r = 0; r < 8; ++r) {
      float s0 = msk0 ? -1e30f : sa0[r];
      float s1 = msk1 ? -1e30f : sa1[r];
      float mx = rmax16(fmaxf(s0, s1));
      float mnew = fmaxf(mrun[r], mx);
      float corr = __expf(mrun[r] - mnew);
      mrun[r] = mnew;
      float p0 = __expf(s0 - mnew);
      float p1 = __expf(s1 - mnew);
      lrun[r] = lrun[r] * corr + rsum16(p0 + p1);
#pragma unroll
      for (int nt = 0; nt < 4; ++nt) oacc[nt][r] *= corr;
      int m = r + half * 8;
      pw[m * LDP + l16]      = f2bf(p0);
      pw[m * LDP + 16 + l16] = f2bf(p1);
    }
    Frag fp;
    {
      const u16* pp = &pw[l16 * LDP + half * 8];
      fp.h[0] = *(const v8bf*)pp;
      fp.h[1] = *(const v8bf*)(pp + 16);
    }
#pragma unroll
    for (int nt = 0; nt < 4; ++nt) {
      Frag fv;
      const u16* vp = vT + ((size_t)((b * H + h) * D) + nt * 16 + l16) * Sp + kc + half * 8;
      fv.h[0] = *(const v8bf*)vp;
      fv.h[1] = *(const v8bf*)(vp + 16);
      oacc[nt] = wmma_bf16(fp.v, fv.v, oacc[nt]);
    }
  }

#pragma unroll
  for (int r = 0; r < 8; ++r) {
    int sq = q0 + r + half * 8;
    if (sq < S) {
      float inv = 1.0f / lrun[r];
      size_t orow = (size_t)(b * S + sq) * C + h * D;
#pragma unroll
      for (int nt = 0; nt < 4; ++nt)
        o[orow + nt * 16 + l16] = f2bf(oacc[nt][r] * inv);
    }
  }
}

// ================= host orchestration =================
extern "C" void kernel_launch(void* const* d_in, const int* in_sizes, int n_in,
                              void* d_out, int out_size, void* d_ws, size_t ws_size,
                              hipStream_t stream) {
  (void)in_sizes; (void)n_in; (void)out_size; (void)ws_size;
  const float* x    = (const float*)d_in[0];
  const float* Wq   = (const float*)d_in[1];
  const float* bq   = (const float*)d_in[2];
  const float* Wk   = (const float*)d_in[3];
  const float* Wv   = (const float*)d_in[4];
  const float* bv   = (const float*)d_in[5];
  const float* Wo   = (const float*)d_in[6];
  const float* bo   = (const float*)d_in[7];
  const float* lnaw = (const float*)d_in[8];
  const float* lnab = (const float*)d_in[9];
  const float* W1   = (const float*)d_in[10];
  const float* b1   = (const float*)d_in[11];
  const float* W2   = (const float*)d_in[12];
  const float* b2   = (const float*)d_in[13];
  const float* lnmw = (const float*)d_in[14];
  const float* lnmb = (const float*)d_in[15];
  const float* Wds  = (const float*)d_in[16];
  const float* bds  = (const float*)d_in[17];
  const float* Wus  = (const float*)d_in[18];
  const float* bus  = (const float*)d_in[19];

  const int B = 4, S = 1500, C = 1024, H = 16, D = 64, F = 4096, A = 256;
  const int M = B * S;   // 6000
  const int Mp = 6016;   // padded rows for q/k (flash tail reads)
  const int Sp = 1504;   // padded seq for vT
  const float scale = 0.35355339059327373f;  // 64^-0.25

  char* p = (char*)d_ws;
  auto take = [&](size_t bytes) -> char* {
    char* r = p;
    p += (bytes + 255) & ~(size_t)255;
    return r;
  };
  u16* wWq  = (u16*)take((size_t)C * C * 2);
  u16* wWk  = (u16*)take((size_t)C * C * 2);
  u16* wWv  = (u16*)take((size_t)C * C * 2);
  u16* wWo  = (u16*)take((size_t)C * C * 2);
  u16* wW1  = (u16*)take((size_t)F * C * 2);
  u16* wW2  = (u16*)take((size_t)C * F * 2);
  u16* wWds = (u16*)take((size_t)A * C * 2);
  u16* wWus = (u16*)take((size_t)C * A * 2);
  u16* hB   = (u16*)take((size_t)M * C * 2);  // h1 -> h2 -> x2(bf16), sequential reuse
  size_t phaseA = ((size_t)Mp * C) * 2 + (size_t)M * C + (size_t)B * H * D * Sp + (size_t)M * C;
  size_t phaseB = (size_t)M * F + (size_t)M * A;
  size_t arenaElems = phaseA > phaseB ? phaseA : phaseB;
  char* arena = take(arenaElems * 2);
  u16* qB  = (u16*)arena;
  u16* kB  = qB + (size_t)Mp * C;
  u16* vB  = kB + (size_t)Mp * C;
  u16* vTB = vB + (size_t)M * C;
  u16* oB  = vTB + (size_t)B * H * D * Sp;
  u16* mB  = (u16*)arena;               // phase B overlay (q/k/v/vT/o dead)
  u16* aB  = mB + (size_t)M * F;
  float* x1 = (float*)take((size_t)M * C * 4);
  float* out = (float*)d_out;

  auto cvt = [&](const float* s, u16* d, size_t n) {
    cvt_bf16_kernel<<<(unsigned)((n + 255) / 256), 256, 0, stream>>>(s, d, n);
  };
  auto gemm = [&](const u16* Am, const u16* Wm, const float* bias, const float* resid,
                  float* oF, u16* oBf, int Mm, int Nn, int Kk, float alpha, int act) {
    dim3 g((unsigned)(Nn / 128), (unsigned)((Mm + 127) / 128));
    gemm_bf16_kernel<<<g, 256, 0, stream>>>(Am, Wm, bias, resid, oF, oBf, Mm, Nn, Kk, alpha, act);
  };

  // 0) weights -> bf16
  cvt(Wq, wWq, (size_t)C * C);
  cvt(Wk, wWk, (size_t)C * C);
  cvt(Wv, wWv, (size_t)C * C);
  cvt(Wo, wWo, (size_t)C * C);
  cvt(W1, wW1, (size_t)F * C);
  cvt(W2, wW2, (size_t)C * F);
  cvt(Wds, wWds, (size_t)A * C);
  cvt(Wus, wWus, (size_t)C * A);

  // 1) h1 = LN_attn(x)
  ln_bf16_kernel<<<M, 256, 0, stream>>>(x, lnaw, lnab, hB, C);

  // 2) q = (h1 Wq^T + bq)*scale ; k = (h1 Wk^T)*scale ; v = h1 Wv^T + bv
  gemm(hB, wWq, bq, nullptr, nullptr, qB, M, C, C, scale, 0);
  gemm(hB, wWk, nullptr, nullptr, nullptr, kB, M, C, C, scale, 0);
  gemm(hB, wWv, bv, nullptr, nullptr, vB, M, C, C, 1.0f, 0);

  // 3) vT[b,h,d,s]
  {
    size_t total = (size_t)B * H * D * Sp;
    vtrans_kernel<<<(unsigned)((total + 255) / 256), 256, 0, stream>>>(vB, vTB, B, S, H, D, Sp);
  }

  // 4) flash attention -> o
  {
    int nQT = (S + 15) / 16;
    dim3 g((unsigned)((nQT + 7) / 8), (unsigned)H, (unsigned)B);
    flash_attn_kernel<<<g, 256, 0, stream>>>(qB, kB, vTB, oB, B, S, H, D, Sp);
  }

  // 5) x1 = x + o Wo^T + bo
  gemm(oB, wWo, bo, x, x1, nullptr, M, C, C, 1.0f, 0);

  // 6) h2 = LN_mlp(x1)
  ln_bf16_kernel<<<M, 256, 0, stream>>>(x1, lnmw, lnmb, hB, C);

  // 7) m = gelu(h2 W1^T + b1)
  gemm(hB, wW1, b1, nullptr, nullptr, mB, M, F, C, 1.0f, 1);

  // 8) x2 = x1 + m W2^T + b2 -> d_out (fp32) + bf16 copy into hB
  gemm(mB, wW2, b2, x1, out, hB, M, C, F, 1.0f, 0);

  // 9) a = gelu(x2 Wds^T + bds)
  gemm(hB, wWds, bds, nullptr, nullptr, aB, M, A, C, 1.0f, 1);

  // 10) out = x2 + a Wus^T + bus (in-place residual on d_out)
  gemm(aB, wWus, bus, out, out, nullptr, M, C, A, 1.0f, 0);
}
